// MultiHeadAttention_8220567404861
// MI455X (gfx1250) — compile-verified, matrix operand order checked by reading, not run on hardware
//
#include <hip/hip_runtime.h>
#include <math.h>

// ---------------------------------------------------------------------------
// MHA forward for MI455X (gfx1250): f16 WMMA everywhere, f32 accum/softmax.
// B=2, S=2048, D=1024, H=16, HD=64
// Round 2: TDM (tensor_load_to_lds) double-buffered V staging in PV kernel,
// b128 LDS stores for transpose staging, global_prefetch of next k-tiles.
// ---------------------------------------------------------------------------

typedef _Float16 half_t;
typedef __attribute__((ext_vector_type(16))) _Float16 v16h;
typedef __attribute__((ext_vector_type(8)))  _Float16 v8h;
typedef __attribute__((ext_vector_type(8)))  float    v8f;
typedef __attribute__((ext_vector_type(4)))  float    v4f;
typedef __attribute__((ext_vector_type(4)))  unsigned int u32x4;
typedef __attribute__((ext_vector_type(8)))  int      i32x8;
typedef __attribute__((ext_vector_type(4)))  int      i32x4;

union V16 { v16h v; v8h h[2]; };

static constexpr int Bc = 2, Sc = 2048, Dc = 1024, Hc = 16, HDc = 64;
static constexpr float SCALEc = 0.125f;   // 64^-0.5
static constexpr int LDK = 40;            // LDS row stride in halves (80B = 20 DW)

#if defined(__gfx1250__) && __has_builtin(__builtin_amdgcn_tensor_load_to_lds) && \
    __has_builtin(__builtin_amdgcn_s_wait_tensorcnt)
#define USE_TDM 1
#else
#define USE_TDM 0
#endif

__device__ __forceinline__ v8f wmma16(v16h a, v16h b, v8f c) {
  return __builtin_amdgcn_wmma_f32_16x16x32_f16(false, a, false, b, (short)0, c,
                                                false, false);
}

__device__ __forceinline__ v16h ldfrag(const half_t* base, int kb) {
  V16 u;
  u.h[0] = *(const v8h*)(base + kb);
  u.h[1] = *(const v8h*)(base + 16 + kb);
  return u.v;
}

__device__ __forceinline__ v8h cvt8(v4f a, v4f b) {
  v8h r;
  r[0] = (half_t)a[0]; r[1] = (half_t)a[1]; r[2] = (half_t)a[2]; r[3] = (half_t)a[3];
  r[4] = (half_t)b[0]; r[5] = (half_t)b[1]; r[6] = (half_t)b[2]; r[7] = (half_t)b[3];
  return r;
}

#if USE_TDM
// Flat LDS address low 32 bits == LDS byte offset (aperture in [63:32]).
__device__ __forceinline__ unsigned int lds_off_of(const void* p) {
  return (unsigned int)(uintptr_t)p;
}

// TDM: load a 2D f16 tile (tile_dim0=32 elems contiguous, tile_dim1=64 rows,
// row stride = Sc elems) into LDS, padding 4 DWORDs every 16 DWORDs so the
// LDS row stride becomes LDK=40 halves.
__device__ __forceinline__ void tdm_load_tile(const half_t* gsrc,
                                              unsigned int lds_byte_off) {
  unsigned long long ga = (unsigned long long)(uintptr_t)gsrc;
  u32x4 g0;
  g0[0] = 1u;                                             // count=1, user mode
  g0[1] = lds_byte_off;                                   // lds_addr
  g0[2] = (unsigned int)ga;                               // global_addr[31:0]
  g0[3] = (unsigned int)((ga >> 32) & 0x01FFFFFFu) | 0x80000000u; // [56:32]|type=2
  i32x8 g1;
  g1[0] = (int)((1u << 16)        // data_size = 2 bytes
              | (1u << 20)        // pad_enable
              | (3u << 22)        // pad_interval: every 16 DWORDs (64B)
              | (3u << 25));      // pad_amount: 4 DWORDs
  g1[1] = (int)(((unsigned int)Sc & 0xFFFFu) << 16);      // tensor_dim0 lo16
  g1[2] = (int)(((unsigned int)Sc >> 16) | ((unsigned int)HDc << 16)); // |dim1 lo
  g1[3] = (int)(32u << 16);                               // tile_dim0 = 32
  g1[4] = (int)64;                                        // tile_dim1 = 64
  g1[5] = (int)Sc;                                        // tensor_dim0_stride
  g1[6] = 0;
  g1[7] = 0;
  i32x4 z4 = {0, 0, 0, 0};
#if defined(__clang_major__) && (__clang_major__ >= 23)
  i32x8 z8 = {0, 0, 0, 0, 0, 0, 0, 0};
  __builtin_amdgcn_tensor_load_to_lds(g0, g1, z4, z4, z8, 0);
#else
  __builtin_amdgcn_tensor_load_to_lds(g0, g1, z4, z4, 0);
#endif
}
#endif  // USE_TDM

// ---------------------------------------------------------------------------
// Kernel 1: fused QKV projection.  X[BS,1024] @ W[1024,1024] + bias.
// Grid: (BS/64, 3072/64).  Q,K -> [B,H,S,HD] f16;  V -> [B,H,HD,S] f16.
// ---------------------------------------------------------------------------
__global__ __launch_bounds__(256) void qkv_kernel(
    const float* __restrict__ q_in, const float* __restrict__ k_in,
    const float* __restrict__ v_in,
    const float* __restrict__ Wq, const float* __restrict__ bq,
    const float* __restrict__ Wk, const float* __restrict__ bk,
    const float* __restrict__ Wv, const float* __restrict__ bv,
    half_t* __restrict__ qb, half_t* __restrict__ kb_, half_t* __restrict__ vtb) {
  __shared__ half_t lds_a[64 * LDK];
  __shared__ half_t lds_b[64 * LDK];

  const int tid  = threadIdx.x;
  const int wave = tid >> 5, lane = tid & 31;
  const int m0 = blockIdx.x * 64;          // row tile in [0, B*S)
  const int n0 = blockIdx.y * 64;          // col tile in [0, 3072)
  const int which = n0 >> 10;              // 0=Q 1=K 2=V
  const int nc0 = n0 & 1023;

  const float* X    = (which == 0) ? q_in : (which == 1) ? k_in : v_in;
  const float* W    = (which == 0) ? Wq   : (which == 1) ? Wk   : Wv;
  const float* bias = (which == 0) ? bq   : (which == 1) ? bk   : bv;

  const int wm = wave >> 1;                // 0..3 M subtile
  const int wn = wave & 1;                 // 0..1 N subtile (32 wide)
  const int lm = lane & 15;
  const int kb = (lane >> 4) * 8;
  const int mofs = (lane >> 4) * 8;

  // staging coordinates
  const int ar = tid >> 2, ac = (tid & 3) * 8;     // A: row, k-chunk
  const int bn = tid & 63, bk8 = (tid >> 6) * 8;   // B: n-col, k-chunk

  v8f acc0 = {}, acc1 = {};

  for (int kk = 0; kk < Dc; kk += 32) {
    { // stage A (64x32 f32 -> f16), one b128 LDS store per thread
      const float* src = X + (size_t)(m0 + ar) * Dc + kk + ac;
      v4f f0 = *(const v4f*)src;
      v4f f1 = *(const v4f*)(src + 4);
      *(v8h*)(lds_a + ar * LDK + ac) = cvt8(f0, f1);
    }
    { // stage B transposed: gather a k-column of W for fixed n; one b128 store
      const float* src = W + (size_t)(kk + bk8) * Dc + nc0 + bn;
      v8h hb;
#pragma unroll
      for (int i = 0; i < 8; ++i) hb[i] = (half_t)src[(size_t)i * Dc];
      *(v8h*)(lds_b + bn * LDK + bk8) = hb;
    }
    if (kk + 32 < Dc) {  // prefetch next k-tile -> global_prefetch_b8
      __builtin_prefetch(X + (size_t)(m0 + ar) * Dc + kk + 32 + ac, 0, 0);
      __builtin_prefetch(W + (size_t)(kk + 32 + bk8) * Dc + nc0 + bn, 0, 0);
    }
    __syncthreads();
    v16h a  = ldfrag(lds_a + (wm * 16 + lm) * LDK, kb);
    v16h b0 = ldfrag(lds_b + (wn * 32 + lm) * LDK, kb);
    v16h b1 = ldfrag(lds_b + (wn * 32 + 16 + lm) * LDK, kb);
    acc0 = wmma16(a, b0, acc0);
    acc1 = wmma16(a, b1, acc1);
    __syncthreads();
  }

#pragma unroll
  for (int sub = 0; sub < 2; ++sub) {
    v8f acc = sub ? acc1 : acc0;
    int n = nc0 + wn * 32 + sub * 16 + lm;
    float bv_ = bias[n];
    int h = n >> 6, hd = n & 63;
#pragma unroll
    for (int r = 0; r < 8; ++r) {
      int row = m0 + wm * 16 + mofs + r;    // b*S + s
      int bi = row >> 11, s = row & (Sc - 1);
      half_t hv = (half_t)(acc[r] + bv_);
      if (which == 2) {
        vtb[(((size_t)bi * Hc + h) * HDc + hd) * Sc + s] = hv;
      } else {
        half_t* dst = (which == 0) ? qb : kb_;
        dst[(((size_t)bi * Hc + h) * Sc + s) * HDc + hd] = hv;
      }
    }
  }
}

// ---------------------------------------------------------------------------
// Kernel 2: scores + softmax.  One block = 16 query rows of one (b,h).
// Each wave owns key tiles j = wave + 8*t -> 16x2048 scores in registers.
// ---------------------------------------------------------------------------
__global__ __launch_bounds__(256) void attn_softmax_kernel(
    const half_t* __restrict__ qb, const half_t* __restrict__ kbuf,
    float* __restrict__ attn_w) {
  __shared__ float red[8][16];
  __shared__ float rowmax_s[16];
  __shared__ float rowsum_s[16];

  const int blk = blockIdx.x;
  const int bh  = blk >> 7;
  const int sq0 = (blk & 127) * 16;
  const half_t* Q = qb   + (size_t)bh * Sc * HDc;
  const half_t* K = kbuf + (size_t)bh * Sc * HDc;

  const int wave = threadIdx.x >> 5, lane = threadIdx.x & 31;
  const int lm = lane & 15;
  const int kb = (lane >> 4) * 8;
  const int mofs = (lane >> 4) * 8;

  const half_t* qrow = Q + (size_t)(sq0 + lm) * HDc;
  v16h aq0 = ldfrag(qrow, kb);
  v16h aq1 = ldfrag(qrow + 32, kb);

  v8f acc[16];
#pragma unroll
  for (int t = 0; t < 16; ++t) {
    int j = wave + (t << 3);
    const half_t* krow = K + (size_t)(j * 16 + lm) * HDc;
    v16h bk0 = ldfrag(krow, kb);
    v16h bk1 = ldfrag(krow + 32, kb);
    v8f c = {};
    c = wmma16(aq0, bk0, c);
    c = wmma16(aq1, bk1, c);
#pragma unroll
    for (int r = 0; r < 8; ++r) c[r] *= SCALEc;
    acc[t] = c;
  }

  // row max over 2048 keys
  float pmax[8];
#pragma unroll
  for (int r = 0; r < 8; ++r) pmax[r] = acc[0][r];
#pragma unroll
  for (int t = 1; t < 16; ++t)
#pragma unroll
    for (int r = 0; r < 8; ++r) pmax[r] = fmaxf(pmax[r], acc[t][r]);
#pragma unroll
  for (int mask = 1; mask < 16; mask <<= 1)
#pragma unroll
    for (int r = 0; r < 8; ++r)
      pmax[r] = fmaxf(pmax[r], __shfl_xor(pmax[r], mask, 32));
  if (lm == 0) {
#pragma unroll
    for (int r = 0; r < 8; ++r) red[wave][mofs + r] = pmax[r];
  }
  __syncthreads();
  if (threadIdx.x < 16) {
    float m = red[0][threadIdx.x];
#pragma unroll
    for (int w = 1; w < 8; ++w) m = fmaxf(m, red[w][threadIdx.x]);
    rowmax_s[threadIdx.x] = m;
  }
  __syncthreads();
  float rmax[8];
#pragma unroll
  for (int r = 0; r < 8; ++r) rmax[r] = rowmax_s[mofs + r];

  // exp + row sum
  float psum[8];
#pragma unroll
  for (int r = 0; r < 8; ++r) psum[r] = 0.0f;
#pragma unroll
  for (int t = 0; t < 16; ++t)
#pragma unroll
    for (int r = 0; r < 8; ++r) {
      float p = __expf(acc[t][r] - rmax[r]);
      acc[t][r] = p;
      psum[r] += p;
    }
#pragma unroll
  for (int mask = 1; mask < 16; mask <<= 1)
#pragma unroll
    for (int r = 0; r < 8; ++r) psum[r] += __shfl_xor(psum[r], mask, 32);
  if (lm == 0) {
#pragma unroll
    for (int r = 0; r < 8; ++r) red[wave][mofs + r] = psum[r];
  }
  __syncthreads();
  if (threadIdx.x < 16) {
    float s = red[0][threadIdx.x];
#pragma unroll
    for (int w = 1; w < 8; ++w) s += red[w][threadIdx.x];
    rowsum_s[threadIdx.x] = s;
  }
  __syncthreads();
  float rinv[8];
#pragma unroll
  for (int r = 0; r < 8; ++r) rinv[r] = 1.0f / rowsum_s[mofs + r];

  float* wout = attn_w + ((size_t)bh * Sc + sq0) * Sc;
#pragma unroll
  for (int t = 0; t < 16; ++t) {
    int key = (wave + (t << 3)) * 16 + lm;
#pragma unroll
    for (int r = 0; r < 8; ++r)
      wout[(size_t)(mofs + r) * Sc + key] = acc[t][r] * rinv[r];
  }
}

// ---------------------------------------------------------------------------
// Kernel 3: O_head = P @ V.  P read back as f32 from d_out; V f16 [hd][s].
// V tile staged to LDS by the Tensor Data Mover, double-buffered; wave 0
// issues TDM + s_wait_tensorcnt, workgroup barrier publishes the buffer.
// ---------------------------------------------------------------------------
__global__ __launch_bounds__(256) void pv_kernel(
    const float* __restrict__ attn_w, const half_t* __restrict__ vtb,
    half_t* __restrict__ attn_out) {
  __shared__ half_t lds_v[2][64 * LDK];

  const int bh = blockIdx.x;
  const int b = bh >> 4, h = bh & 15;
  const float*  P  = attn_w + (size_t)bh * Sc * Sc;
  const half_t* Vt = vtb    + (size_t)bh * HDc * Sc;

  const int tid = threadIdx.x;
  const int wave = tid >> 5, lane = tid & 31;
  const int lm = lane & 15;
  const int kb = (lane >> 4) * 8;
  const int mofs = (lane >> 4) * 8;
  const int sq = blockIdx.y * 128 + wave * 16;

  v8f acc[4] = {v8f{}, v8f{}, v8f{}, v8f{}};

  constexpr int nIter = Sc / 32;
#if USE_TDM
  if (wave == 0) tdm_load_tile(Vt, lds_off_of(&lds_v[0][0]));
#endif
  for (int it = 0; it < nIter; ++it) {
    const int kk = it * 32;
    const int buf = it & 1;
#if USE_TDM
    if (wave == 0) {
      if (it + 1 < nIter) {
        // buffer buf^1 was last read before the previous end-of-iter barrier
        tdm_load_tile(Vt + (size_t)(kk + 32), lds_off_of(&lds_v[buf ^ 1][0]));
        __builtin_amdgcn_s_wait_tensorcnt((short)1);  // current tile landed
      } else {
        __builtin_amdgcn_s_wait_tensorcnt((short)0);
      }
    }
#else
    { // cooperative staging fallback
      int hd = tid >> 2, c = (tid & 3) * 8;
      *(v8h*)(&lds_v[buf][hd * LDK + c]) =
          *(const v8h*)(Vt + (size_t)hd * Sc + kk + c);
    }
#endif
    __syncthreads();

    // A fragment: probs f32 -> f16
    const float* prow = P + (size_t)(sq + lm) * Sc + kk;
    v4f f0 = *(const v4f*)(prow + kb);
    v4f f1 = *(const v4f*)(prow + kb + 4);
    v4f f2 = *(const v4f*)(prow + 16 + kb);
    v4f f3 = *(const v4f*)(prow + 16 + kb + 4);
    V16 ua;
    ua.h[0] = cvt8(f0, f1);
    ua.h[1] = cvt8(f2, f3);
#pragma unroll
    for (int sub = 0; sub < 4; ++sub) {
      v16h bf = ldfrag(&lds_v[buf][(sub * 16 + lm) * LDK], kb);
      acc[sub] = wmma16(ua.v, bf, acc[sub]);
    }
    __syncthreads();  // all reads of lds_v[buf] done before TDM refills it
  }

#pragma unroll
  for (int sub = 0; sub < 4; ++sub) {
    int hd = sub * 16 + lm;
#pragma unroll
    for (int r = 0; r < 8; ++r) {
      int s = sq + mofs + r;
      attn_out[(size_t)(b * Sc + s) * (Hc * HDc) + h * HDc + hd] =
          (half_t)acc[sub][r];
    }
  }
}

// ---------------------------------------------------------------------------
// Kernel 4: final projection.  attn_out[BS,1024](f16) @ Wo[1024,1024] + bo.
// ---------------------------------------------------------------------------
__global__ __launch_bounds__(256) void oproj_kernel(
    const half_t* __restrict__ attn_out, const float* __restrict__ Wo,
    const float* __restrict__ bo, float* __restrict__ out) {
  __shared__ half_t lds_b[64 * LDK];

  const int tid = threadIdx.x;
  const int wave = tid >> 5, lane = tid & 31;
  const int m0 = blockIdx.x * 64;
  const int n0 = blockIdx.y * 64;
  const int wm = wave >> 1, wn = wave & 1;
  const int lm = lane & 15;
  const int kb = (lane >> 4) * 8;
  const int mofs = (lane >> 4) * 8;
  const int bn = tid & 63, bk8 = (tid >> 6) * 8;

  v8f acc0 = {}, acc1 = {};

  for (int kk = 0; kk < Dc; kk += 32) {
    { // stage Wo transposed: gather k-column, one b128 LDS store
      const float* src = Wo + (size_t)(kk + bk8) * Dc + n0 + bn;
      v8h hb;
#pragma unroll
      for (int i = 0; i < 8; ++i) hb[i] = (half_t)src[(size_t)i * Dc];
      *(v8h*)(lds_b + bn * LDK + bk8) = hb;
    }
    if (kk + 32 < Dc) {
      __builtin_prefetch(Wo + (size_t)(kk + 32 + bk8) * Dc + n0 + bn, 0, 0);
      __builtin_prefetch(
          attn_out + (size_t)(m0 + wm * 16 + lm) * Dc + kk + 32 + kb, 0, 0);
    }
    __syncthreads();
    const half_t* arow = attn_out + (size_t)(m0 + wm * 16 + lm) * Dc + kk;
    V16 ua;
    ua.h[0] = *(const v8h*)(arow + kb);
    ua.h[1] = *(const v8h*)(arow + 16 + kb);
    v16h b0 = ldfrag(lds_b + (wn * 32 + lm) * LDK, kb);
    v16h b1 = ldfrag(lds_b + (wn * 32 + 16 + lm) * LDK, kb);
    acc0 = wmma16(ua.v, b0, acc0);
    acc1 = wmma16(ua.v, b1, acc1);
    __syncthreads();
  }

#pragma unroll
  for (int sub = 0; sub < 2; ++sub) {
    v8f acc = sub ? acc1 : acc0;
    int n = n0 + wn * 32 + sub * 16 + lm;
    float bb = bo[n];
#pragma unroll
    for (int r = 0; r < 8; ++r) {
      int row = m0 + wm * 16 + mofs + r;
      out[(size_t)row * Dc + n] = acc[r] + bb;
    }
  }
}

// ---------------------------------------------------------------------------
extern "C" void kernel_launch(void* const* d_in, const int* in_sizes, int n_in,
                              void* d_out, int out_size, void* d_ws, size_t ws_size,
                              hipStream_t stream) {
  const float* q_in = (const float*)d_in[0];
  const float* k_in = (const float*)d_in[1];
  const float* v_in = (const float*)d_in[2];
  const float* Wq = (const float*)d_in[3];
  const float* bq = (const float*)d_in[4];
  const float* Wk = (const float*)d_in[5];
  const float* bk = (const float*)d_in[6];
  const float* Wv = (const float*)d_in[7];
  const float* bv = (const float*)d_in[8];
  const float* Wo = (const float*)d_in[9];
  const float* bo = (const float*)d_in[10];

  float* out_main = (float*)d_out;                          // [B,S,D]
  float* attn_w   = (float*)d_out + (size_t)Bc * Sc * Dc;   // [B,H,S,S]

  const size_t NQKV = (size_t)Bc * Hc * Sc * HDc;
  char* ws = (char*)d_ws;
  half_t* qb       = (half_t*)(ws);
  half_t* kbuf     = (half_t*)(ws + NQKV * sizeof(half_t));
  half_t* vtb      = (half_t*)(ws + 2 * NQKV * sizeof(half_t));
  half_t* attn_out = (half_t*)(ws + 3 * NQKV * sizeof(half_t));
  (void)in_sizes; (void)n_in; (void)out_size; (void)ws_size;

  dim3 blk(256);
  qkv_kernel<<<dim3((Bc * Sc) / 64, 3072 / 64), blk, 0, stream>>>(
      q_in, k_in, v_in, Wq, bq, Wk, bk, Wv, bv, qb, kbuf, vtb);
  attn_softmax_kernel<<<dim3(Bc * Hc * (Sc / 16)), blk, 0, stream>>>(
      qb, kbuf, attn_w);
  pv_kernel<<<dim3(Bc * Hc, Sc / 128), blk, 0, stream>>>(attn_w, vtb, attn_out);
  oproj_kernel<<<dim3((Bc * Sc) / 64, Dc / 64), blk, 0, stream>>>(
      attn_out, Wo, bo, out_main);
}
